// RVQBottleneck_23957327577859
// MI455X (gfx1250) — matrix operand order repaired and checked by reading, not compiled, not run on hardware
//
#include <hip/hip_runtime.h>
#include <hip/hip_bf16.h>

// ---- problem constants (from reference setup_inputs) ----
#define B_    16
#define C_    64
#define N_    4096
#define Q_    8
#define K_    8192
#define PTILE 128          // points per workgroup
#define WAVES 4            // 128 threads / wave32
#define WPTS  32           // points per wave (two 16x16 WMMA sub-tiles)
#define KT    (K_ / 16)    // 512 code tiles of 16
#define RPITCH 65          // residual LDS pitch (bank-conflict pad)

static constexpr size_t OUT_ELEMS  = (size_t)B_ * C_ * N_;   // 4,194,304
static constexpr size_t IDX_OFF    = OUT_ELEMS;              // indices region
static constexpr size_t IDX_ELEMS  = (size_t)B_ * N_ * Q_;   // 524,288
static constexpr size_t LOSS_OFF   = IDX_OFF + IDX_ELEMS;    // scalar loss
static constexpr int    NPARTIAL   = (B_ * N_ / PTILE) * WAVES; // 2048 loss partials

typedef __attribute__((ext_vector_type(16))) __bf16 v16bf;
typedef __attribute__((ext_vector_type(8)))  float  v8f;

union FragU { v16bf v; uint4 u[2]; };
union PackU { __bf16 h[8]; uint4 u; };

__device__ __forceinline__ v8f wmma_bf16(v16bf a, v16bf b, v8f c) {
  // D = A(16x32 bf16) * B(32x16 bf16) + C(16x16 f32)
  return __builtin_amdgcn_wmma_f32_16x16x32_bf16(
      /*neg_a=*/false, a, /*neg_b=*/false, b,
      /*c_mod=*/(short)0, c, /*reuse_a=*/false, /*reuse_b=*/false);
}

// ---- kernel 1: nh[q][k] = -0.5 * |codebook[q][k]|^2 ----
__global__ void rvq_cbsq(const float* __restrict__ cb, float* __restrict__ nh) {
  int qk = blockIdx.x * blockDim.x + threadIdx.x;       // < Q_*K_
  const float4* p = (const float4*)(cb + (size_t)qk * C_);
  float s = 0.f;
#pragma unroll
  for (int j = 0; j < C_ / 4; ++j) {
    float4 v = p[j];
    s += v.x * v.x + v.y * v.y + v.z * v.z + v.w * v.w;
  }
  nh[qk] = -0.5f * s;
}

// ---- kernel 2: main RVQ ----
__launch_bounds__(PTILE, 1)
__global__ void rvq_main(const float* __restrict__ x,
                         const float* __restrict__ cbs,
                         const float* __restrict__ nhsq,
                         float* __restrict__ out,
                         float* __restrict__ wsloss) {
  __shared__ __align__(16) float  lds_r[PTILE][RPITCH];     // residuals (f32)
  __shared__ __align__(16) __bf16 cb_hi[2][16][C_];         // staged code tile, hi part
  __shared__ __align__(16) __bf16 cb_lo[2][16][C_];         // staged code tile, lo part
  __shared__ int bidx_lds[PTILE];

  const int t    = threadIdx.x;
  const int w    = t >> 5;
  const int lane = t & 31;
  const int half = lane >> 4;        // 0: lanes 0-15, 1: lanes 16-31
  const int nn   = lane & 15;        // column (code-within-tile) per C/D layout
  const int p0   = blockIdx.x * PTILE;
  const int b    = p0 >> 12;         // N_ = 4096
  const int n0   = p0 & (N_ - 1);
  const float* xb = x + (size_t)b * C_ * N_ + n0;

  // staging assignment: thread t loads 8 floats of the 16x64 code tile
  const int srow = t >> 3;           // 0..15
  const int scol = (t & 7) * 8;      // 0,8,...,56  (16B-aligned bf16 packs)

  // init residual = x tile (transpose [C][n] -> [point][C]); coalesced per c
  for (int c = 0; c < C_; ++c) lds_r[t][c] = xb[(size_t)c * N_ + t];

  auto stage = [&](int buf, float4 f0, float4 f1) {
    float fv[8] = {f0.x, f0.y, f0.z, f0.w, f1.x, f1.y, f1.z, f1.w};
    PackU ph, pl;
#pragma unroll
    for (int j = 0; j < 8; ++j) {
      __bf16 hb = (__bf16)fv[j];
      ph.h[j] = hb;
      pl.h[j] = (__bf16)(fv[j] - (float)hb);
    }
    *(uint4*)&cb_hi[buf][srow][scol] = ph.u;
    *(uint4*)&cb_lo[buf][srow][scol] = pl.u;
  };

  float lossAcc = 0.f;

  for (int q = 0; q < Q_; ++q) {
    const float* cbq = cbs + (size_t)q * K_ * C_;
    const float* nhq = nhsq + q * K_;
    __syncthreads();  // residual stable (init or previous update)

    // ---- build A fragments (residual, bf16 hi/lo split) : contiguous K (k = 16*half + e), same as B ----
    v16bf aH[2][2], aL[2][2];   // [subtile][C-chunk of 32]
#pragma unroll
    for (int s = 0; s < 2; ++s) {
      const int row = w * WPTS + s * 16 + nn;   // A: lane%16 -> M
#pragma unroll
      for (int ch = 0; ch < 2; ++ch) {
#pragma unroll
        for (int e = 0; e < 16; ++e) {
          int kl = e + 16 * half;  // contiguous K layout (matches B)
          float rv = lds_r[row][ch * 32 + kl];
          __bf16 hb = (__bf16)rv;
          aH[s][ch][e] = hb;
          aL[s][ch][e] = (__bf16)(rv - (float)hb);
        }
      }
    }

    float best0[8], best1[8];
    int   bi0[8],  bi1[8];
#pragma unroll
    for (int r = 0; r < 8; ++r) {
      best0[r] = -3.4e38f; best1[r] = -3.4e38f; bi0[r] = 0; bi1[r] = 0;
    }

    // prologue: stage code tile 0
    {
      const float4* g = (const float4*)(cbq + (size_t)srow * C_ + scol);
      stage(0, g[0], g[1]);
    }
    __syncthreads();

#pragma unroll 2
    for (int kt = 0; kt < KT; ++kt) {
      const int buf = kt & 1;
      const bool pref = (kt + 1 < KT);
      float4 f0, f1;
      if (pref) {
        const float4* g =
            (const float4*)(cbq + ((size_t)(kt + 1) * 16 + srow) * C_ + scol);
        f0 = g[0]; f1 = g[1];
      }

      const float nh = nhq[kt * 16 + nn];   // -0.5*|cb|^2 for this lane's code

      // ---- B fragments: lanes 0-15 hold K=0..15, lanes 16-31 K=16..31 (seq) ----
      v16bf bH[2], bL[2];
#pragma unroll
      for (int ch = 0; ch < 2; ++ch) {
        const uint4* ph = (const uint4*)&cb_hi[buf][nn][ch * 32 + half * 16];
        const uint4* pl = (const uint4*)&cb_lo[buf][nn][ch * 32 + half * 16];
        FragU fh, fl;
        fh.u[0] = ph[0]; fh.u[1] = ph[1];
        fl.u[0] = pl[0]; fl.u[1] = pl[1];
        bH[ch] = fh.v;   bL[ch] = fl.v;
      }

      // score = dot(r, cb) - 0.5*|cb|^2  (argmax == argmin distance)
      v8f acc0 = {nh, nh, nh, nh, nh, nh, nh, nh};
      v8f acc1 = acc0;
      // hi*hi + hi*lo + lo*hi over C=64 (two 32-chunks) — 6 WMMAs per sub-tile
      acc0 = wmma_bf16(aH[0][0], bH[0], acc0); acc0 = wmma_bf16(aH[0][1], bH[1], acc0);
      acc1 = wmma_bf16(aH[1][0], bH[0], acc1); acc1 = wmma_bf16(aH[1][1], bH[1], acc1);
      acc0 = wmma_bf16(aH[0][0], bL[0], acc0); acc0 = wmma_bf16(aH[0][1], bL[1], acc0);
      acc1 = wmma_bf16(aH[1][0], bL[0], acc1); acc1 = wmma_bf16(aH[1][1], bL[1], acc1);
      acc0 = wmma_bf16(aL[0][0], bH[0], acc0); acc0 = wmma_bf16(aL[0][1], bH[1], acc0);
      acc1 = wmma_bf16(aL[1][0], bH[0], acc1); acc1 = wmma_bf16(aL[1][1], bH[1], acc1);

      const int kidx = kt * 16 + nn;
#pragma unroll
      for (int r = 0; r < 8; ++r) {
        if (acc0[r] > best0[r]) { best0[r] = acc0[r]; bi0[r] = kidx; }
        if (acc1[r] > best1[r]) { best1[r] = acc1[r]; bi1[r] = kidx; }
      }

      if (pref) stage(buf ^ 1, f0, f1);
      __syncthreads();
    }

    // ---- cross-lane argmax within each 16-lane half (same M rows, different n) ----
#pragma unroll
    for (int r = 0; r < 8; ++r) {
#pragma unroll
      for (int off = 1; off < 16; off <<= 1) {
        float od = __shfl_xor(best0[r], off, 32);
        int   oi = __shfl_xor(bi0[r],  off, 32);
        if (od > best0[r] || (od == best0[r] && oi < bi0[r])) { best0[r] = od; bi0[r] = oi; }
        od = __shfl_xor(best1[r], off, 32);
        oi = __shfl_xor(bi1[r],  off, 32);
        if (od > best1[r] || (od == best1[r] && oi < bi1[r])) { best1[r] = od; bi1[r] = oi; }
      }
    }
    if (nn == 0) {   // lanes 0 and 16 publish winners; C/D map: M = r + 8*half
#pragma unroll
      for (int r = 0; r < 8; ++r) {
        int m0 = w * WPTS +       r + 8 * half;
        int m1 = w * WPTS + 16 +  r + 8 * half;
        bidx_lds[m0] = bi0[r];
        bidx_lds[m1] = bi1[r];
        out[IDX_OFF + (size_t)(p0 + m0) * Q_ + q] = (float)bi0[r];
        out[IDX_OFF + (size_t)(p0 + m1) * Q_ + q] = (float)bi1[r];
      }
    }
    __syncthreads();

    // ---- residual update + commitment-loss accumulation (wave-local points) ----
    for (int e = lane; e < WPTS * C_; e += 32) {
      int il = e >> 6, c = e & 63;
      int i  = w * WPTS + il;
      int idx = bidx_lds[i];
      float cv = cbq[(size_t)idx * C_ + c];
      float rn = lds_r[i][c] - cv;
      lds_r[i][c] = rn;
      lossAcc += rn * rn;
    }
  }
  __syncthreads();

  // quantized = x - residual_final ; transpose back to [B,C,N]
  for (int c = 0; c < C_; ++c)
    out[(size_t)b * C_ * N_ + (size_t)c * N_ + n0 + t] =
        xb[(size_t)c * N_ + t] - lds_r[t][c];

  // deterministic per-wave loss partial
#pragma unroll
  for (int off = 1; off < 32; off <<= 1) lossAcc += __shfl_xor(lossAcc, off, 32);
  if (lane == 0) wsloss[blockIdx.x * WAVES + w] = lossAcc;
}

// ---- kernel 3: fixed-order loss reduction (replay-deterministic) ----
__global__ void rvq_loss_reduce(const float* __restrict__ wsloss,
                                float* __restrict__ out) {
  __shared__ float sm[256];
  int t = threadIdx.x;
  float s = 0.f;
  for (int j = t; j < NPARTIAL; j += 256) s += wsloss[j];
  sm[t] = s;
  __syncthreads();
  for (int st = 128; st > 0; st >>= 1) {
    if (t < st) sm[t] += sm[t + st];
    __syncthreads();
  }
  if (t == 0)
    out[LOSS_OFF] = sm[0] * (1.0f / ((float)Q_ * (float)B_ * (float)N_ * (float)C_));
}

extern "C" void kernel_launch(void* const* d_in, const int* in_sizes, int n_in,
                              void* d_out, int out_size, void* d_ws, size_t ws_size,
                              hipStream_t stream) {
  const float* x   = (const float*)d_in[0];   // [B, C, N] f32
  const float* cbs = (const float*)d_in[1];   // [Q, K, C] f32
  float* out = (float*)d_out;                 // out | indices | loss (concat)
  float* ws  = (float*)d_ws;
  float* nhsq   = ws;            // Q*K floats: -0.5*|cb|^2
  float* wsloss = ws + Q_ * K_;  // NPARTIAL floats: per-wave loss partials

  rvq_cbsq<<<(Q_ * K_) / 256, 256, 0, stream>>>(cbs, nhsq);
  rvq_main<<<(B_ * N_) / PTILE, PTILE, 0, stream>>>(x, cbs, nhsq, out, wsloss);
  rvq_loss_reduce<<<1, 256, 0, stream>>>(wsloss, out);
}
